// Attention_21646635172219
// MI455X (gfx1250) — matrix operand order repaired and checked by reading, not compiled, not run on hardware
//
#include <hip/hip_runtime.h>

// Sizes from the reference
#define BSZ 32
#define SSZ 4096
#define DSZ 512
#define ATILE 128          // S-rows per workgroup in the main kernel
#define ASTRIDE 520        // LDS row stride in bf16 elements (1040B = 260 dwords, 260%64=4 -> conflict-free)
#define SMB_DW 4096        // dwords per B staging buffer (16 frags * 256 dwords)

typedef __attribute__((ext_vector_type(16))) __bf16 v16bf;
typedef __attribute__((ext_vector_type(8)))  float  v8f;
typedef __attribute__((ext_vector_type(8)))  unsigned int v8u;
typedef __attribute__((ext_vector_type(4)))  int v4i;

#if defined(__AMDGCN__) && __has_builtin(__builtin_amdgcn_global_load_async_to_lds_b128)
#define USE_ASYNC 1
#else
#define USE_ASYNC 0
#endif

#if USE_ASYNC
typedef __attribute__((address_space(1))) v4i* g_v4i_p;
typedef __attribute__((address_space(3))) v4i* l_v4i_p;
__device__ __forceinline__ void async_b128(const unsigned int* g, unsigned int* l) {
    __builtin_amdgcn_global_load_async_to_lds_b128(
        (g_v4i_p)(g), (l_v4i_p)(l), 0, 0);
}
__device__ __forceinline__ void wait_asynccnt0() {
#if __has_builtin(__builtin_amdgcn_s_wait_asynccnt)
    __builtin_amdgcn_s_wait_asynccnt(0);
#else
    asm volatile("s_wait_asynccnt 0x0" ::: "memory");
#endif
}
#endif

__device__ __forceinline__ unsigned short f2bf(float f) {
    unsigned u = __builtin_bit_cast(unsigned, f);
    unsigned r = u + 0x7FFFu + ((u >> 16) & 1u);   // round-to-nearest-even
    return (unsigned short)(r >> 16);
}

// ---------------------------------------------------------------------------
// Repack W2 (f32 [D,D], out[s,e]=sum_d v[s,d]*W2[d,e]) into bf16 B-matrix
// fragments. Fragment f = nt*16+kt covers N=nt*16..+15, K=kt*32..+31.
// B 32x16 in this code's contiguous k order (A uses the same): lane L -> column n=L%16, K-half = L/16; dword d holds
// K = half*16 + 2d, 2d+1. Stored frag-major:
//   dword offset = f*256 + (d/4)*128 + lane*4 + (d%4)
// ---------------------------------------------------------------------------
__global__ void k_pack_w2(const float* __restrict__ W2, unsigned int* __restrict__ Bp) {
    int id   = blockIdx.x * 256 + threadIdx.x;     // 0..131071 dwords
    int f    = id >> 8;
    int rem  = id & 255;
    int p    = rem >> 7;
    int lane = (rem >> 2) & 31;
    int dq   = rem & 3;
    int d    = p * 4 + dq;
    int nt   = f >> 4, kt = f & 15;
    int n    = nt * 16 + (lane & 15);
    int k    = kt * 32 + (lane >> 4) * 16 + d * 2;
    unsigned lo = f2bf(W2[k * DSZ + n]);
    unsigned hi = f2bf(W2[(k + 1) * DSZ + n]);
    Bp[id] = lo | (hi << 16);
}

// ---------------------------------------------------------------------------
// qpb[b,n] = (query @ W1)[b,n] + b1[n] + b2[n]   (bv dropped: softmax shift-inv)
// ---------------------------------------------------------------------------
__global__ void k_qproj(const float* __restrict__ q, const float* __restrict__ W1,
                        const float* __restrict__ b1, const float* __restrict__ b2,
                        float* __restrict__ qpb) {
    int tid = blockIdx.x * 256 + threadIdx.x;      // 0..16383
    int b = tid >> 9, n = tid & 511;
    float acc = b1[n] + b2[n];
    const float* qr = q + b * DSZ;
    for (int d = 0; d < DSZ; ++d) acc += qr[d] * W1[d * DSZ + n];
    qpb[tid] = acc;
}

// ---------------------------------------------------------------------------
// Main fused kernel: per (s-tile, batch) workgroup:
//  * stage 128x512 values tile -> LDS bf16 (coalesced), hoist per-wave A
//    fragments into 128 VGPRs,
//  * per N-tile: B fragments staged once per WG into LDS (all 8 waves share),
//    double-buffered across nt via global_load_async_to_lds_b128 (ASYNCcnt),
//  * inner loop: sched_group_barrier-enforced pipeline DSx4,(WMMA,DSx2)x14,
//    WMMAx2 so each v_wmma overlaps the next fragment's ds_load_b128 pair,
//  * fused tanh(qpb+vproj)*V with 16-lane shfl_xor reduction into scores.
// ---------------------------------------------------------------------------
__global__ __launch_bounds__(256) void k_score(
        const float* __restrict__ values, const unsigned int* __restrict__ Bp,
        const float* __restrict__ qpb,    const float* __restrict__ Vvec,
        float* __restrict__ scores) {
    extern __shared__ unsigned short smA[];                 // ATILE*ASTRIDE bf16
    unsigned int* smB = (unsigned int*)(smA + ATILE * ASTRIDE); // 2 * SMB_DW dwords
    int b      = blockIdx.y;
    int s_base = blockIdx.x * ATILE;
    int t      = threadIdx.x;

    // ---- stage values tile: 128 rows x 512 f32 -> bf16 LDS ----
    const float4* src = (const float4*)(values + ((size_t)b * SSZ + s_base) * DSZ);
    #pragma unroll 4
    for (int i = 0; i < 64; ++i) {
        int idx = i * 256 + t;
        int row = idx >> 7, c4 = idx & 127;
        float4 v = src[row * 128 + c4];
        unsigned d0 = (unsigned)f2bf(v.x) | ((unsigned)f2bf(v.y) << 16);
        unsigned d1 = (unsigned)f2bf(v.z) | ((unsigned)f2bf(v.w) << 16);
        *(uint2*)(smA + row * ASTRIDE + c4 * 4) = make_uint2(d0, d1);
    }

    // ---- stage B fragments for nt=0 (16KB, whole workgroup) ----
#if USE_ASYNC
    #pragma unroll
    for (int i = 0; i < 4; ++i) {
        int dw = (i * 256 + t) * 4;
        async_b128(Bp + dw, smB + dw);
    }
    wait_asynccnt0();
#else
    #pragma unroll
    for (int i = 0; i < 4; ++i) {
        uint4 v = ((const uint4*)Bp)[i * 256 + t];
        ((uint4*)smB)[i * 256 + t] = v;
    }
#endif
    __syncthreads();

    int wave = t >> 5, lane = t & 31;
    int hf = lane >> 4, ln = lane & 15;
    int m0 = wave * 16;

    // ---- hoist this wave's 16 A-fragments into registers (read LDS once) ----
    const unsigned short* arow = smA + (m0 + ln) * ASTRIDE;
    v8u a[16];
    #pragma unroll
    for (int kt = 0; kt < 16; ++kt) {
        int k0 = kt * 32 + hf * 16;
        uint4 a0 = *(const uint4*)(arow + k0);
        uint4 a1 = *(const uint4*)(arow + k0 + 8);
        a[kt] = (v8u){a0.x, a0.y, a0.z, a0.w, a1.x, a1.y, a1.z, a1.w};
    }

    float sacc[8];
    #pragma unroll
    for (int r = 0; r < 8; ++r) sacc[r] = 0.f;

    int buf = 0;
    #pragma unroll 1
    for (int nt = 0; nt < 32; ++nt) {
        float qv = qpb[b * DSZ + nt * 16 + ln];
        float Vv = Vvec[nt * 16 + ln];

        // issue next nt's B staging so it overlaps this nt's 16 WMMAs
#if USE_ASYNC
        if (nt < 31) {
            const unsigned int* srcn = Bp + (nt + 1) * SMB_DW;
            unsigned int* dstn = smB + (buf ^ 1) * SMB_DW;
            #pragma unroll
            for (int i = 0; i < 4; ++i) {
                int dw = (i * 256 + t) * 4;
                async_b128(srcn + dw, dstn + dw);
            }
        }
#else
        uint4 stg[4];
        if (nt < 31) {
            const uint4* srcn = (const uint4*)(Bp + (nt + 1) * SMB_DW);
            #pragma unroll
            for (int i = 0; i < 4; ++i) stg[i] = srcn[i * 256 + t];
        }
#endif

        // ---- 16 WMMAs over K=512 from the LDS-staged B fragments ----
        v8f c = {};
        const uint4* bq = (const uint4*)(smB + buf * SMB_DW); // frag kt at uint4 idx kt*64
        #pragma unroll
        for (int kt = 0; kt < 16; ++kt) {
            uint4 b0  = bq[kt * 64 + lane];
            uint4 b1v = bq[kt * 64 + 32 + lane];
            v8u bu = (v8u){b0.x, b0.y, b0.z, b0.w, b1v.x, b1v.y, b1v.z, b1v.w};
            c = __builtin_amdgcn_wmma_f32_16x16x32_bf16(
                    false, __builtin_bit_cast(v16bf, a[kt]),
                    false, __builtin_bit_cast(v16bf, bu),
                    (short)0, c, false, false);
        }
#if defined(__AMDGCN__) && __has_builtin(__builtin_amdgcn_sched_group_barrier)
        // Pin the pipeline: 4 ds-reads ahead, then (1 wmma, 2 ds-reads)*14,
        // then the last 2 wmmas. Each wmma overlaps the next fragment's loads.
        __builtin_amdgcn_sched_group_barrier(0x100, 4, 0);   // DS read x4
        #pragma unroll
        for (int i = 0; i < 14; ++i) {
            __builtin_amdgcn_sched_group_barrier(0x008, 1, 0); // WMMA x1
            __builtin_amdgcn_sched_group_barrier(0x100, 2, 0); // DS read x2
        }
        __builtin_amdgcn_sched_group_barrier(0x008, 2, 0);   // WMMA x2
#endif

        // fuse: hidden = tanh(qpb + vproj); score += hidden . V over these 16 n's
        #pragma unroll
        for (int r = 0; r < 8; ++r) {
            float sp = tanhf(qv + c[r]) * Vv;
            sp += __shfl_xor(sp, 1, 32);
            sp += __shfl_xor(sp, 2, 32);
            sp += __shfl_xor(sp, 4, 32);
            sp += __shfl_xor(sp, 8, 32);           // reduce within 16-lane half
            sacc[r] += sp;
        }

        // complete the nt+1 staging, then rotate buffers
#if USE_ASYNC
        if (nt < 31) wait_asynccnt0();
#else
        if (nt < 31) {
            uint4* dstn = (uint4*)(smB + (buf ^ 1) * SMB_DW);
            #pragma unroll
            for (int i = 0; i < 4; ++i) dstn[i * 256 + t] = stg[i];
        }
#endif
        __syncthreads();
        buf ^= 1;
    }

    if (ln == 0) {
        #pragma unroll
        for (int r = 0; r < 8; ++r) {
            int m = s_base + m0 + hf * 8 + r;       // C/D layout: M = r + 8*half
            scores[b * SSZ + m] = sacc[r];
        }
    }
}

// ---------------------------------------------------------------------------
// Softmax over S per batch, in place in the attn region of d_out.
// ---------------------------------------------------------------------------
__global__ void k_softmax(float* __restrict__ attn) {
    __shared__ float red[256];
    int b = blockIdx.x, t = threadIdx.x;
    float* s = attn + b * SSZ;
    float v[16];
    float mx = -3.4e38f;
    #pragma unroll
    for (int i = 0; i < 16; ++i) { v[i] = s[i * 256 + t]; mx = fmaxf(mx, v[i]); }
    red[t] = mx; __syncthreads();
    for (int o = 128; o > 0; o >>= 1) { if (t < o) red[t] = fmaxf(red[t], red[t + o]); __syncthreads(); }
    mx = red[0]; __syncthreads();
    float sum = 0.f;
    #pragma unroll
    for (int i = 0; i < 16; ++i) { v[i] = __expf(v[i] - mx); sum += v[i]; }
    red[t] = sum; __syncthreads();
    for (int o = 128; o > 0; o >>= 1) { if (t < o) red[t] += red[t + o]; __syncthreads(); }
    float inv = 1.0f / red[0];
    #pragma unroll
    for (int i = 0; i < 16; ++i) s[i * 256 + t] = v[i] * inv;
}

// ---------------------------------------------------------------------------
// context: split-S partials (deterministic, no fp atomics), then tree reduce.
// ---------------------------------------------------------------------------
__global__ void k_ctx_part(const float* __restrict__ values, const float* __restrict__ attn,
                           float* __restrict__ ctxp) {
    int b = blockIdx.y, ch = blockIdx.x, t = threadIdx.x;
    int d0 = t * 2;
    float a0 = 0.f, a1 = 0.f;
    const float* vp = values + ((size_t)b * SSZ + ch * 256) * DSZ;
    const float* ap = attn + b * SSZ + ch * 256;
    for (int s = 0; s < 256; ++s) {
        float w = ap[s];                            // uniform -> scalar load
        float2 vv = *(const float2*)(vp + (size_t)s * DSZ + d0);
        a0 += w * vv.x; a1 += w * vv.y;
    }
    float* out = ctxp + (b * 16 + ch) * DSZ + d0;
    out[0] = a0; out[1] = a1;
}

__global__ void k_ctx_reduce(const float* __restrict__ ctxp, float* __restrict__ ctx) {
    int tid = blockIdx.x * 256 + threadIdx.x;      // 0..16383
    int b = tid >> 9, d = tid & 511;
    float acc = 0.f;
    #pragma unroll
    for (int c = 0; c < 16; ++c) acc += ctxp[(b * 16 + c) * DSZ + d];
    ctx[tid] = acc;
}

extern "C" void kernel_launch(void* const* d_in, const int* in_sizes, int n_in,
                              void* d_out, int out_size, void* d_ws, size_t ws_size,
                              hipStream_t stream) {
    const float* query  = (const float*)d_in[0];
    const float* values = (const float*)d_in[1];
    const float* W1     = (const float*)d_in[2];
    const float* b1     = (const float*)d_in[3];
    const float* W2     = (const float*)d_in[4];
    const float* b2     = (const float*)d_in[5];
    const float* Vvec   = (const float*)d_in[6];
    // d_in[7] = bv: softmax is shift-invariant, drop it.

    // workspace layout
    float*        qpb  = (float*)d_ws;                                   // 64 KB
    unsigned int* Bp   = (unsigned int*)((char*)d_ws + 65536);           // 512 KB
    float*        ctxp = (float*)((char*)d_ws + 65536 + 524288);         // 1 MB

    float* ctx  = (float*)d_out;             // [B, D]
    float* attn = (float*)d_out + BSZ * DSZ; // [B, S]

    size_t smbytes = (size_t)ATILE * ASTRIDE * 2 + (size_t)2 * SMB_DW * 4;

    k_pack_w2<<<512, 256, 0, stream>>>(W2, Bp);
    k_qproj<<<64, 256, 0, stream>>>(query, W1, b1, b2, qpb);
    k_score<<<dim3(SSZ / ATILE, BSZ), 256, smbytes, stream>>>(values, Bp, qpb, Vvec, attn);
    k_softmax<<<BSZ, 256, 0, stream>>>(attn);
    k_ctx_part<<<dim3(16, BSZ), 256, 0, stream>>>(values, attn, ctxp);
    k_ctx_reduce<<<64, 256, 0, stream>>>(ctxp, ctx);
}
